// MyMsa_68186900791588
// MI455X (gfx1250) — matrix operand order repaired and checked by reading, not compiled, not run on hardware
//
#include <hip/hip_runtime.h>
#include <hip/hip_bf16.h>

#define DEV __device__ __forceinline__

typedef __attribute__((ext_vector_type(16))) __bf16 v16bf;
typedef __attribute__((ext_vector_type(8)))  float  v8f;

namespace {
constexpr int B = 8, S = 1024, D = 1024, H = 16, DH = 64;

// workspace layout, in bf16 (u16) element offsets
constexpr size_t OFF_WQT = 0;                            // [3][H][e][d] (q,k,v transposed)
constexpr size_t OFF_WOT = OFF_WQT + (size_t)3*H*DH*DH;  // [e][d] Wo transposed
constexpr size_t OFF_Q   = OFF_WOT + (size_t)D*D;        // [B][H][S][DH]
constexpr size_t OFF_K   = OFF_Q  + (size_t)B*H*S*DH;    // [B][H][S][DH]
constexpr size_t OFF_V   = OFF_K  + (size_t)B*H*S*DH;    // [B][H][DH][S]  (transposed!)
constexpr size_t OFF_AO  = OFF_V  + (size_t)B*H*S*DH;    // attention out, [B][S][D]
} // namespace

DEV unsigned short f2bf(float x) {           // round-to-nearest-even fp32 -> bf16
  unsigned int u = __float_as_uint(x);
  u += 0x7FFFu + ((u >> 16) & 1u);
  return (unsigned short)(u >> 16);
}

union BF16Frag { v16bf v; unsigned short u[16]; unsigned int d[8]; };

// A fragment (16xK tile, row-major bf16, element (m, k0+kk)); ISA 16-bit A layout:
// lanes 0-15: M=lane, VGPR r -> K pairs {0,1..6,7 | 16,17..22,23}; lanes 16-31: +8.
DEV v16bf frag_a_bf16(const unsigned short* buf, int stride, int k0, int lane) {
  BF16Frag f; int m = lane & 15, hi = lane >> 4;
#pragma unroll
  for (int r = 0; r < 8; ++r) {
    int kb = ((r < 4) ? 2 * r : 16 + 2 * (r - 4)) + 8 * hi + k0;
    f.d[r] = *(const unsigned int*)(buf + (size_t)m * stride + kb);
  }
  return f.v;
}

// B fragment (Kx16). bufT is N-major / K-contiguous: element (k, n) = bufT[n*stride + k].
// same K order as A: lanes 0-15: VGPR r -> K pairs {0,1..6,7 | 16,17..22,23}; lanes 16-31: +8.
DEV v16bf frag_b_bf16(const unsigned short* bufT, int stride, int k0, int lane) {
  BF16Frag f; int n = lane & 15, hi = lane >> 4;
#pragma unroll
  for (int r = 0; r < 8; ++r) {
    int kb = ((r < 4) ? 2 * r : 16 + 2 * (r - 4)) + 8 * hi + k0;
    f.d[r] = *(const unsigned int*)(bufT + (size_t)n * stride + kb);
  }
  return f.v;
}

// A fragment gathered straight from fp32 memory (converted on the fly)
DEV v16bf frag_a_f32(const float* buf, int stride, int k0, int lane) {
  BF16Frag f; int m = lane & 15, hi = lane >> 4;
#pragma unroll
  for (int r = 0; r < 8; ++r) {
    int kb = ((r < 4) ? 2 * r : 16 + 2 * (r - 4)) + 8 * hi + k0;
    const float* p = buf + (size_t)m * stride + kb;
    f.u[2 * r]     = f2bf(p[0]);
    f.u[2 * r + 1] = f2bf(p[1]);
  }
  return f.v;
}

DEV v8f wmma_bf16(v16bf a, v16bf b, v8f c) {
  // (neg_a, A, neg_b, B, c_mod, C, reuse_a, reuse_b)
  return __builtin_amdgcn_wmma_f32_16x16x32_bf16(false, a, false, b, (short)0, c,
                                                 false, false);
}

// --- CDNA5 async HBM->LDS copy path (ASYNCcnt-tracked) ----------------------
DEV unsigned lds_off32(const void* p) {       // generic LDS ptr: low dword = offset
  return (unsigned)(unsigned long long)p;
}
DEV void async_ld_b128(unsigned lds_byte_off, unsigned long long gaddr) {
  asm volatile("global_load_async_to_lds_b128 %0, %1, off"
               :: "v"(lds_byte_off), "v"(gaddr) : "memory");
}
DEV void wait_asynccnt0() { asm volatile("s_wait_asynccnt 0x0" ::: "memory"); }

// ---------------------------------------------------------------------------
// Kernel 0: convert weights to bf16, transposed (so B-fragments are K-contiguous)
// ---------------------------------------------------------------------------
__global__ void msa_wconvert(const float* Wq, const float* Wk, const float* Wv,
                             const float* Wo, unsigned short* ws) {
  size_t i = (size_t)blockIdx.x * blockDim.x + threadIdx.x;
  constexpr size_t NW = (size_t)3 * H * DH * DH;
  if (i < NW) {
    int which = (int)(i / ((size_t)H * DH * DH));
    int rem   = (int)(i % ((size_t)H * DH * DH));
    int h = rem / (DH * DH);
    int e = (rem / DH) % DH;
    int d = rem % DH;
    const float* src = (which == 0) ? Wq : ((which == 1) ? Wk : Wv);
    ws[OFF_WQT + i] = f2bf(src[((size_t)h * DH + d) * DH + e]);
  } else if (i < NW + (size_t)D * D) {
    size_t j = i - NW;
    int e = (int)(j / D), d = (int)(j % D);
    ws[OFF_WOT + j] = f2bf(Wo[(size_t)d * D + e]);
  }
}

// ---------------------------------------------------------------------------
// Kernel 1: fused per-head Q/K/V projection. One wave = one 16x64 tile of
// q, k, or v for one (b, h). 8 x v_wmma_f32_16x16x32_bf16 per wave.
// V is written TRANSPOSED [B][H][DH][S] so the attention kernel can async-DMA
// its [d][key] LDS tile with no transpose work in the hot loop.
// ---------------------------------------------------------------------------
__global__ void msa_qkv(const float* __restrict__ seq, const float* __restrict__ bq,
                        const float* __restrict__ bk, const float* __restrict__ bv,
                        unsigned short* ws) {
  int lane = threadIdx.x & 31;
  int gw = blockIdx.x * (blockDim.x >> 5) + (threadIdx.x >> 5);
  int s16   = gw & 63;
  int h     = (gw >> 6) & 15;
  int b     = (gw >> 10) & 7;
  int which = gw >> 13;            // 0=q, 1=k, 2=v
  int s0 = s16 * 16;

  const float* xa = seq + ((size_t)b * S + s0) * D + h * DH;  // 16 x 64, stride D
  v16bf a0 = frag_a_f32(xa, D, 0, lane);
  v16bf a1 = frag_a_f32(xa, D, 32, lane);

  const unsigned short* wt = ws + OFF_WQT + ((size_t)which * H + h) * DH * DH; // [e][d]
  const float* bias = (which == 0) ? bq : ((which == 1) ? bk : bv);
  unsigned short* out = ws + OFF_Q + (size_t)which * B * H * S * DH
                      + ((size_t)b * H + h) * S * DH;

  int n = lane & 15, hi = lane >> 4;
#pragma unroll
  for (int j = 0; j < 4; ++j) {
    float b0 = bias[(size_t)h * DH + 16 * j + n];    // bias broadcast along rows
    v8f c = {b0, b0, b0, b0, b0, b0, b0, b0};
    v16bf w0 = frag_b_bf16(wt + (size_t)(16 * j) * DH, DH, 0, lane);
    v16bf w1 = frag_b_bf16(wt + (size_t)(16 * j) * DH, DH, 32, lane);
    c = wmma_bf16(a0, w0, c);
    c = wmma_bf16(a1, w1, c);
#pragma unroll
    for (int r = 0; r < 8; ++r) {
      int m = r + 8 * hi;                            // D-matrix row mapping
      if (which == 2)                                // v: [d][s] transposed store
        out[(size_t)(16 * j + n) * S + (s0 + m)] = f2bf(c[r]);
      else                                           // q,k: [s][d]
        out[(size_t)(s0 + m) * DH + 16 * j + n] = f2bf(c[r]);
    }
  }
}

// ---------------------------------------------------------------------------
// Kernel 2: flash attention. Block = (b, h, 128 query rows); wave owns 16 rows.
// K/V tiles of 32 keys double-buffered in LDS via GLOBAL_LOAD_ASYNC_TO_LDS_B128
// (ASYNCcnt), overlapping HBM->LDS DMA for tile t+1 with WMMA on tile t.
// Online softmax; P re-laid-out through per-wave LDS scratch. 256 WMMAs/wave.
// ---------------------------------------------------------------------------
__global__ void msa_attn(unsigned short* ws) {
  __shared__ unsigned short Kbuf[2][32 * 64];   // [key][d]   (N-major for QK^T B-frag)
  __shared__ unsigned short Vbuf[2][64 * 32];   // [d][key]   (N-major for PV  B-frag)
  __shared__ unsigned short Ps[8][16 * 32];     // per-wave P tile [m][key]

  int tid = threadIdx.x;
  int lane = tid & 31, w = tid >> 5;
  int qt = blockIdx.x & 7;
  int h  = (blockIdx.x >> 3) & 15;
  int b  = blockIdx.x >> 7;
  int s0 = qt * 128 + w * 16;

  const unsigned short* qp = ws + OFF_Q + (((size_t)b * H + h) * S + s0) * DH;
  const unsigned short* kp = ws + OFF_K + ((size_t)b * H + h) * S * DH;   // [key][d]
  const unsigned short* vT = ws + OFF_V + ((size_t)b * H + h) * DH * S;   // [d][key]

  v16bf a0 = frag_a_bf16(qp, DH, 0, lane);    // q tile lives in regs all iterations
  v16bf a1 = frag_a_bf16(qp, DH, 32, lane);

  const v8f zero = {0.f, 0.f, 0.f, 0.f, 0.f, 0.f, 0.f, 0.f};
  float mi[8], li[8];
  v8f acc[4];
#pragma unroll
  for (int r = 0; r < 8; ++r) { mi[r] = -3.0e38f; li[r] = 0.f; }
#pragma unroll
  for (int j = 0; j < 4; ++j) acc[j] = zero;

  const float inv_scale = 1.0f / 4096.0f;     // reference uses 1/DH^2
  int n = lane & 15, hi = lane >> 4;
  int vrow = tid >> 2, vchunk = tid & 3;      // V-tile DMA mapping: 64 rows x 64B

  // issue 2 x async B128 per thread: K tile (4KB contiguous) + V tile (64x64B rows)
  auto stage = [&](int buf, int t0) {
    async_ld_b128(lds_off32(&Kbuf[buf][0]) + (unsigned)tid * 16u,
                  (unsigned long long)(kp + (size_t)t0 * DH) +
                      (unsigned long long)tid * 16u);
    async_ld_b128(lds_off32(&Vbuf[buf][0]) + (unsigned)(vrow * 64 + vchunk * 16),
                  (unsigned long long)(vT + (size_t)vrow * S + t0) +
                      (unsigned long long)vchunk * 16u);
  };

  constexpr int NT = S / 32;
  stage(0, 0);                       // prologue: fill buffer 0
  wait_asynccnt0();
  __syncthreads();

  for (int kt = 0; kt < NT; ++kt) {
    int cur = kt & 1;
    if (kt + 1 < NT) stage(cur ^ 1, (kt + 1) * 32);   // DMA next tile, no wait

    const unsigned short* Kt = &Kbuf[cur][0];
    const unsigned short* Vt = &Vbuf[cur][0];

    // scores S[16x32] = q . k^T  (B(k=d, n=key) = Kt[key][d])
    v8f sc0 = zero, sc1 = zero;
    sc0 = wmma_bf16(a0, frag_b_bf16(Kt, DH, 0, lane), sc0);
    sc0 = wmma_bf16(a1, frag_b_bf16(Kt, DH, 32, lane), sc0);
    sc1 = wmma_bf16(a0, frag_b_bf16(Kt + 16 * DH, DH, 0, lane), sc1);
    sc1 = wmma_bf16(a1, frag_b_bf16(Kt + 16 * DH, DH, 32, lane), sc1);

    // online softmax (rows live across 16-lane halves -> xor-shuffle reductions)
#pragma unroll
    for (int r = 0; r < 8; ++r) {
      float x0 = sc0[r] * inv_scale, x1 = sc1[r] * inv_scale;
      float vmax = fmaxf(x0, x1);
      vmax = fmaxf(vmax, __shfl_xor(vmax, 1, 32));
      vmax = fmaxf(vmax, __shfl_xor(vmax, 2, 32));
      vmax = fmaxf(vmax, __shfl_xor(vmax, 4, 32));
      vmax = fmaxf(vmax, __shfl_xor(vmax, 8, 32));
      float mn = fmaxf(mi[r], vmax);
      float alpha = __expf(mi[r] - mn);
      mi[r] = mn;
      float p0 = __expf(x0 - mn);
      float p1 = __expf(x1 - mn);
      float rs = p0 + p1;
      rs += __shfl_xor(rs, 1, 32);
      rs += __shfl_xor(rs, 2, 32);
      rs += __shfl_xor(rs, 4, 32);
      rs += __shfl_xor(rs, 8, 32);
      li[r] = li[r] * alpha + rs;
#pragma unroll
      for (int j = 0; j < 4; ++j) acc[j][r] *= alpha;
      int m = r + 8 * hi;
      Ps[w][m * 32 + n]      = f2bf(p0);   // re-layout P: D-layout -> row-major
      Ps[w][m * 32 + 16 + n] = f2bf(p1);
    }
    asm volatile("s_wait_dscnt 0" ::: "memory");   // P stores visible before gather

    // O += P[16x32] . V[32x64]  (B(k=key, n=d) = Vt[d][key])
    v16bf ap = frag_a_bf16(Ps[w], 32, 0, lane);
#pragma unroll
    for (int j = 0; j < 4; ++j)
      acc[j] = wmma_bf16(ap, frag_b_bf16(Vt + 16 * j * 32, 32, 0, lane), acc[j]);

    if (kt + 1 < NT) wait_asynccnt0();   // next tile DMA done (overlapped w/ compute)
    __syncthreads();
  }

  // normalize and scatter into hstacked [B,S,D] bf16 buffer
  unsigned short* op = ws + OFF_AO + ((size_t)b * S + s0) * D + h * DH;
#pragma unroll
  for (int j = 0; j < 4; ++j)
#pragma unroll
    for (int r = 0; r < 8; ++r) {
      int m = r + 8 * hi;
      op[(size_t)m * D + 16 * j + n] = f2bf(acc[j][r] / li[r]);
    }
}

// ---------------------------------------------------------------------------
// Kernel 3: output projection [B*S,1024] x [1024,1024] + bias. One wave =
// 16x64 output tile, K=1024 in 32 steps -> 128 WMMAs per wave. Wo^T lives in
// L2 (2 MB bf16, reused by every wave).
// ---------------------------------------------------------------------------
__global__ void msa_oproj(const unsigned short* __restrict__ ws,
                          const float* __restrict__ bo, float* __restrict__ out) {
  int lane = threadIdx.x & 31;
  int gw = blockIdx.x * (blockDim.x >> 5) + (threadIdx.x >> 5);
  int et = gw & 15;                 // 64-wide column tile
  size_t row0 = (size_t)(gw >> 4) * 16;

  const unsigned short* ap = ws + OFF_AO + row0 * D;
  const unsigned short* wt = ws + OFF_WOT + (size_t)et * 64 * D;  // [e][d]
  int n = lane & 15, hi = lane >> 4;

  v8f c[4];
#pragma unroll
  for (int j = 0; j < 4; ++j) {
    float b0 = bo[et * 64 + 16 * j + n];
    c[j] = (v8f){b0, b0, b0, b0, b0, b0, b0, b0};
  }

  for (int kk = 0; kk < D / 32; ++kk) {
    int k0 = kk * 32;
    v16bf a = frag_a_bf16(ap, D, k0, lane);
#pragma unroll
    for (int j = 0; j < 4; ++j)
      c[j] = wmma_bf16(a, frag_b_bf16(wt + (size_t)(16 * j) * D, D, k0, lane), c[j]);
  }
#pragma unroll
  for (int j = 0; j < 4; ++j)
#pragma unroll
    for (int r = 0; r < 8; ++r) {
      int m = r + 8 * hi;
      out[(row0 + m) * D + et * 64 + 16 * j + n] = c[j][r];
    }
}

// ---------------------------------------------------------------------------
extern "C" void kernel_launch(void* const* d_in, const int* in_sizes, int n_in,
                              void* d_out, int out_size, void* d_ws, size_t ws_size,
                              hipStream_t stream) {
  const float* seq = (const float*)d_in[0];
  const float* Wq  = (const float*)d_in[1];
  const float* bq  = (const float*)d_in[2];
  const float* Wk  = (const float*)d_in[3];
  const float* bk  = (const float*)d_in[4];
  const float* Wv  = (const float*)d_in[5];
  const float* bv  = (const float*)d_in[6];
  const float* Wo  = (const float*)d_in[7];
  const float* bo  = (const float*)d_in[8];
  unsigned short* ws = (unsigned short*)d_ws;
  float* out = (float*)d_out;

  {
    size_t nconv = (size_t)3 * H * DH * DH + (size_t)D * D;
    msa_wconvert<<<(int)((nconv + 255) / 256), 256, 0, stream>>>(Wq, Wk, Wv, Wo, ws);
  }
  // 3 * B * H * (S/16) waves, 8 waves (256 threads) per block
  msa_qkv<<<3 * B * H * (S / 16) / 8, 256, 0, stream>>>(seq, bq, bk, bv, ws);
  // one block per (b, h, 128-row query tile)
  msa_attn<<<B * H * (S / 128), 256, 0, stream>>>(ws);
  // B*S/16 row tiles x D/64 col tiles, 8 waves per block
  msa_oproj<<<(B * (S / 16)) * (D / 64) / 8, 256, 0, stream>>>(ws, bo, out);
}
